// BalancedHamiltonLayer_70531952935592
// MI455X (gfx1250) — matrix operand order repaired and checked by reading, not compiled, not run on hardware
//
#include <hip/hip_runtime.h>

// MI455X / gfx1250, wave32. The layer collapses to Out(8192x1024) = X * W^T + bias
// with W[(k*64+j),(s*64+i)] = sum_r H[r,k,s]*B_r[j,i] (1024x1024, precomputed).
// GEMM runs on v_wmma_f32_16x16x32_bf16 with split-bf16 3-product accumulation
// (Xhi*Whi + Xhi*Wlo + Xlo*Whi into one f32 C) for ~fp32 accuracy.
//
// Primary path pre-splits X into bf16 hi/lo planes stored in natural row-major
// layout; contiguous K order makes every A/B fragment is ONE contiguous 32B v16bf load ->
// inner loop is pure vmem + wmma. Fallback (small workspace) converts in-loop.

typedef __attribute__((ext_vector_type(16))) __bf16 v16bf;
typedef __attribute__((ext_vector_type(8)))  __bf16 v8bf;
typedef __attribute__((ext_vector_type(8)))  float  v8f;
typedef __attribute__((ext_vector_type(4)))  float  v4f;
typedef __attribute__((ext_vector_type(4)))  unsigned short v4us;

static constexpr int DK = 1024;  // inner dim (s*64+i)
static constexpr int DN = 1024;  // output dim (k*64+j)
static constexpr int DM = 8192;  // rows = B*T

// ---------------- bf16 helpers (bit-exact RNE, pure integer ops) -----------
__device__ __forceinline__ unsigned short f2bf_u(float f) {
  unsigned int u = __builtin_bit_cast(unsigned int, f);
  return (unsigned short)((u + 0x7FFFu + ((u >> 16) & 1u)) >> 16);
}
__device__ __forceinline__ float bfu2f(unsigned short h) {
  unsigned int u = ((unsigned int)h) << 16;
  return __builtin_bit_cast(float, u);
}
__device__ __forceinline__ __bf16 u2bf(unsigned short h) {
  return __builtin_bit_cast(__bf16, h);
}

// ---------------------------------------------------------------------------
// Kernel 1: W[kj][si] = sum_r H[r,k,s] * factors_B[r,j,i], split to bf16 hi/lo.
// Row-major (out=1024, in=1024). Hamilton component/sign from quaternion blocks.
// ---------------------------------------------------------------------------
__global__ __launch_bounds__(256) void build_w_kernel(
    const float* __restrict__ A_stack,    // (8,4,4,4)
    const float* __restrict__ factors_B,  // (8,64,64)
    unsigned short* __restrict__ Whi,
    unsigned short* __restrict__ Wlo) {
  const int idx = blockIdx.x * 256 + threadIdx.x;  // 0 .. 1024*1024-1
  const int kj = idx >> 10, si = idx & 1023;
  const int k = kj >> 6, j = kj & 63;
  const int s = si >> 6, i = si & 63;
  const int a = k >> 2, p = k & 3, b = s >> 2, q = s & 3;

  const int   comp_tab[16] = {0, 1, 2, 3,  1, 0, 3, 2,  2, 3, 0, 1,  3, 2, 1, 0};
  const float sign_tab[16] = {1.f, -1.f, -1.f, -1.f,
                              1.f,  1.f, -1.f,  1.f,
                              1.f,  1.f,  1.f, -1.f,
                              1.f, -1.f,  1.f,  1.f};
  const int   c  = comp_tab[a * 4 + b];
  const float sg = sign_tab[a * 4 + b];

  float acc = 0.f;
#pragma unroll
  for (int r = 0; r < 8; ++r)
    acc = fmaf(A_stack[((r * 4 + c) * 4 + p) * 4 + q],
               factors_B[(r * 64 + j) * 64 + i], acc);
  acc *= sg;

  const unsigned short hi = f2bf_u(acc);
  Whi[idx] = hi;
  Wlo[idx] = f2bf_u(acc - bfu2f(hi));
}

// ---------------------------------------------------------------------------
// Kernel 2: split X (f32) into bf16 hi/lo planes, stored in natural row-major order;
// inside each 32-wide K chunk the four 8-elem groups keep order 0,1,2,3 so
// a lane's v16bf A fragment (K = h*16..h*16+15) is contiguous.
// ---------------------------------------------------------------------------
__global__ __launch_bounds__(256) void split_x_kernel(
    const float* __restrict__ X,
    unsigned short* __restrict__ Xhi,
    unsigned short* __restrict__ Xlo) {
  const size_t base = ((size_t)blockIdx.x * 256 + threadIdx.x) * 4;  // 4 f32 each
  const int col = (int)(base & (size_t)(DK - 1));
  const size_t row_off = base - col;  // row * DK
  const int chunk = col >> 5, within = col & 31;
  const int g  = within >> 3;
  const int gp = g;  // natural group order (contiguous K)
  const int colp = (chunk << 5) + (gp << 3) + (within & 7);

  const v4f x = *(const v4f*)(const void*)(X + base);
  v4us h, l;
#pragma unroll
  for (int e = 0; e < 4; ++e) {
    h[e] = f2bf_u(x[e]);
    l[e] = f2bf_u(x[e] - bfu2f(h[e]));
  }
  *(v4us*)(void*)(Xhi + row_off + colp) = h;
  *(v4us*)(void*)(Xlo + row_off + colp) = l;
}

// ---------------------------------------------------------------------------
// Kernel 3 (primary): GEMM on pre-split bf16 planes. 8 waves per block,
// block tile 128x128, wave tile 32(M)x64(N) -> 2x4 C tiles.
// Every fragment is a single contiguous 32B v16bf load.
// ---------------------------------------------------------------------------
__global__ __launch_bounds__(256) void hamilton_gemm_presplit_kernel(
    const unsigned short* __restrict__ Xhi,  // (8192,1024) row-major bf16
    const unsigned short* __restrict__ Xlo,
    const unsigned short* __restrict__ Whi,  // (1024,1024) row-major bf16
    const unsigned short* __restrict__ Wlo,
    const float* __restrict__ bias,
    float* __restrict__ Out) {
  const int lane  = threadIdx.x & 31;
  const int wave  = threadIdx.x >> 5;
  const int mwave = wave & 3;
  const int nwave = wave >> 2;
  const int half  = lane >> 4;
  const int r16   = lane & 15;

  const int mBase = blockIdx.y * 128 + mwave * 32;
  const int nBase = blockIdx.x * 128 + nwave * 64;

  const int kfrag = half * 16;  // contiguous fragment offset (A and B natural)

  v8f acc[2][4] = {};

  for (int kk = 0; kk < DK; kk += 32) {
    v16bf ahi[2], alo[2];
#pragma unroll
    for (int mt = 0; mt < 2; ++mt) {
      const size_t off = (size_t)(mBase + mt * 16 + r16) * DK + kk + kfrag;
      ahi[mt] = *(const v16bf*)(const void*)(Xhi + off);
      alo[mt] = *(const v16bf*)(const void*)(Xlo + off);
    }
    v16bf bhi[4], blo[4];
#pragma unroll
    for (int nt = 0; nt < 4; ++nt) {
      const size_t off = (size_t)(nBase + nt * 16 + r16) * DK + kk + kfrag;
      bhi[nt] = *(const v16bf*)(const void*)(Whi + off);
      blo[nt] = *(const v16bf*)(const void*)(Wlo + off);
    }
#pragma unroll
    for (int mt = 0; mt < 2; ++mt)
#pragma unroll
      for (int nt = 0; nt < 4; ++nt) {
        acc[mt][nt] = __builtin_amdgcn_wmma_f32_16x16x32_bf16(
            false, ahi[mt], false, bhi[nt], (short)0, acc[mt][nt], false, false);
        acc[mt][nt] = __builtin_amdgcn_wmma_f32_16x16x32_bf16(
            false, ahi[mt], false, blo[nt], (short)0, acc[mt][nt], false, false);
        acc[mt][nt] = __builtin_amdgcn_wmma_f32_16x16x32_bf16(
            false, alo[mt], false, bhi[nt], (short)0, acc[mt][nt], false, false);
      }
  }

#pragma unroll
  for (int nt = 0; nt < 4; ++nt) {
    const int col = nBase + nt * 16 + r16;
    const float bv = bias[col];
#pragma unroll
    for (int mt = 0; mt < 2; ++mt) {
      const int rowb = mBase + mt * 16 + half * 8;
#pragma unroll
      for (int v = 0; v < 8; ++v)
        Out[(size_t)(rowb + v) * DN + col] = acc[mt][nt][v] + bv;
    }
  }
}

// ---------------------------------------------------------------------------
// Kernel 3' (fallback, workspace too small for X planes): convert X in-loop.
// A fragments assembled from f32 loads (contiguous K layout), B fragments are
// direct contiguous v16bf loads.
// ---------------------------------------------------------------------------
__global__ __launch_bounds__(256) void hamilton_gemm_fused_kernel(
    const float* __restrict__ X,
    const unsigned short* __restrict__ Whi,
    const unsigned short* __restrict__ Wlo,
    const float* __restrict__ bias,
    float* __restrict__ Out) {
  const int lane  = threadIdx.x & 31;
  const int wave  = threadIdx.x >> 5;
  const int mwave = wave & 3;
  const int nwave = wave >> 2;
  const int half  = lane >> 4;
  const int r16   = lane & 15;

  const int mBase = blockIdx.y * 128 + mwave * 32;
  const int nBase = blockIdx.x * 128 + nwave * 64;

  const int ka0 = half * 16;       // A elems 0..7  : K = half*16 ..
  const int ka1 = half * 16 + 8;   // A elems 8..15 : K = half*16+8 ..
  const int kb  = half * 16;       // B fragment contiguous

  v8f acc[2][4] = {};

  for (int kk = 0; kk < DK; kk += 32) {
    v16bf ahi[2], alo[2];
#pragma unroll
    for (int mt = 0; mt < 2; ++mt) {
      const float* xr = X + (size_t)(mBase + mt * 16 + r16) * DK + kk;
      const v8f xa = *(const v8f*)(const void*)(xr + ka0);
      const v8f xb = *(const v8f*)(const void*)(xr + ka1);
#pragma unroll
      for (int e = 0; e < 8; ++e) {
        const unsigned short h0 = f2bf_u(xa[e]);
        const unsigned short h1 = f2bf_u(xb[e]);
        ahi[mt][e]     = u2bf(h0);
        ahi[mt][8 + e] = u2bf(h1);
        alo[mt][e]     = u2bf(f2bf_u(xa[e] - bfu2f(h0)));
        alo[mt][8 + e] = u2bf(f2bf_u(xb[e] - bfu2f(h1)));
      }
    }
    v16bf bhi[4], blo[4];
#pragma unroll
    for (int nt = 0; nt < 4; ++nt) {
      const size_t off = (size_t)(nBase + nt * 16 + r16) * DK + kk + kb;
      bhi[nt] = *(const v16bf*)(const void*)(Whi + off);
      blo[nt] = *(const v16bf*)(const void*)(Wlo + off);
    }
#pragma unroll
    for (int mt = 0; mt < 2; ++mt)
#pragma unroll
      for (int nt = 0; nt < 4; ++nt) {
        acc[mt][nt] = __builtin_amdgcn_wmma_f32_16x16x32_bf16(
            false, ahi[mt], false, bhi[nt], (short)0, acc[mt][nt], false, false);
        acc[mt][nt] = __builtin_amdgcn_wmma_f32_16x16x32_bf16(
            false, ahi[mt], false, blo[nt], (short)0, acc[mt][nt], false, false);
        acc[mt][nt] = __builtin_amdgcn_wmma_f32_16x16x32_bf16(
            false, alo[mt], false, bhi[nt], (short)0, acc[mt][nt], false, false);
      }
  }

#pragma unroll
  for (int nt = 0; nt < 4; ++nt) {
    const int col = nBase + nt * 16 + r16;
    const float bv = bias[col];
#pragma unroll
    for (int mt = 0; mt < 2; ++mt) {
      const int rowb = mBase + mt * 16 + half * 8;
#pragma unroll
      for (int v = 0; v < 8; ++v)
        Out[(size_t)(rowb + v) * DN + col] = acc[mt][nt][v] + bv;
    }
  }
}

// ---------------------------------------------------------------------------
extern "C" void kernel_launch(void* const* d_in, const int* in_sizes, int n_in,
                              void* d_out, int out_size, void* d_ws, size_t ws_size,
                              hipStream_t stream) {
  (void)in_sizes; (void)n_in; (void)out_size;

  const float* x        = (const float*)d_in[0];  // (4,2048,1024)
  const float* A_stack  = (const float*)d_in[1];  // (8,4,4,4)
  const float* factorsB = (const float*)d_in[2];  // (8,64,64)
  const float* bias     = (const float*)d_in[3];  // (1024)
  float* out = (float*)d_out;

  const size_t wElems = (size_t)DN * DK;      // 1M
  const size_t xElems = (size_t)DM * DK;      // 8M
  const size_t needSplit = (2 * wElems + 2 * xElems) * sizeof(unsigned short);  // 36 MB

  unsigned short* Whi = (unsigned short*)d_ws;
  unsigned short* Wlo = Whi + wElems;

  build_w_kernel<<<(int)(wElems / 256), 256, 0, stream>>>(A_stack, factorsB, Whi, Wlo);

  dim3 grid(DN / 128, DM / 128);
  if (ws_size >= needSplit) {
    unsigned short* Xhi = Wlo + wElems;
    unsigned short* Xlo = Xhi + xElems;
    split_x_kernel<<<(int)(xElems / 4 / 256), 256, 0, stream>>>(x, Xhi, Xlo);
    hamilton_gemm_presplit_kernel<<<grid, 256, 0, stream>>>(Xhi, Xlo, Whi, Wlo, bias, out);
  } else {
    hamilton_gemm_fused_kernel<<<grid, 256, 0, stream>>>(x, Whi, Wlo, bias, out);
  }
}
